// Feature_encoder_27092653703371
// MI455X (gfx1250) — matrix operand order repaired and checked by reading, not compiled, not run on hardware
//
#include <hip/hip_runtime.h>

// ---------------- problem constants ----------------
#define D_    192
#define NH_   8
#define HD_   24
#define LTOK  11520
#define HW0_  9216
#define HW1_  2304
#define H0_   96
#define W0_   96
#define H1_   48
#define W1_   48
#define BB    2
#define NLAYERS 6

typedef _Float16 v16h __attribute__((ext_vector_type(16)));
typedef float    v8f  __attribute__((ext_vector_type(8)));

// ---------------------------------------------------------------------------
// WMMA GEMM, weight panel staged in LDS as fp16.
//   C[m,n] = sum_k A[m,k] * W'[k,n] + bias[n]
//   transB==0: W'[k,n] = W[k*ldw + n];  transB==1: W'[k,n] = W[n*ldw + k]
// Template NT = full output width N (64/128/192/256).
// Block = 256 threads = 8 waves; block computes 128 rows x NT cols.
// Each wave: one 16-row strip, NT/16 f32 accumulators, one shared A fragment
// per K-step feeding NT/16 back-to-back v_wmma_f32_16x16x32_f16, with B
// fragments software-pipelined one deep so ds_load latency overlaps the XDL
// pipe instead of draining dscnt to 0 before every WMMA.
// Dynamic LDS = K*NT*2 bytes (<= 96KB, well under CDNA5's 320KB/WGP).
// Output row remap: row m -> C + (m/cRowsPerBatch)*cBatchStride
//                              + (m%cRowsPerBatch)*ldc   (plain: cRowsPerBatch=M).
// Requires M%128==0, K%32==0, K*NT%256==0.
// ---------------------------------------------------------------------------
template <int NT>
__global__ __launch_bounds__(256)
void k_gemm(const float* __restrict__ A, int lda,
            const float* __restrict__ W, int ldw, int transB,
            const float* __restrict__ bias,
            float* __restrict__ C, int ldc, int cRowsPerBatch, long long cBatchStride,
            int M, int K, int relu)
{
    extern __shared__ _Float16 ldsW[];   // [K][NT] fp16 weight panel

    const int tid  = threadIdx.x;
    const int wv   = tid >> 5;           // wave 0..7
    const int lane = tid & 31;
    const int mloc = lane & 15;
    const int hiK  = lane >> 4;          // 0: K 0..7/16..23, 1: 8..15/24..31
    const int mb   = blockIdx.x * 128 + wv * 16;

    // ---- cooperative stage: W (fp32, optionally transposed) -> LDS fp16 ----
    const int tot = K * NT;              // multiple of 256 for all our shapes
    if (!transB) {
        for (int i = tid; i < tot; i += 256) {
            int k = i / NT, n = i % NT;
            ldsW[i] = (_Float16)W[(size_t)k * ldw + n];
        }
    } else {
        for (int i = tid; i < tot; i += 256) {
            int k = i / NT, n = i % NT;
            ldsW[i] = (_Float16)W[(size_t)n * ldw + k];
        }
    }
    __syncthreads();

    v8f acc[NT / 16];
#pragma unroll
    for (int nt = 0; nt < NT / 16; ++nt) acc[nt] = (v8f){};

    const float* arow = A + (size_t)(mb + mloc) * lda;

    for (int kb = 0; kb < K; kb += 32) {
        // ---- A fragment: 16x32 f16; lane holds row mloc, K-pairs per VGPR ----
        v16h afrag;
#pragma unroll
        for (int v = 0; v < 8; ++v) {
            int k0 = ((v < 4) ? (2 * v) : (16 + 2 * (v - 4))) + 8 * hiK;
            float2 f2 = *(const float2*)(arow + kb + k0);
            afrag[2 * v]     = (_Float16)f2.x;
            afrag[2 * v + 1] = (_Float16)f2.y;
        }
        // WGP-scope prefetch of next A chunk (same wave consumes it next iter)
        __builtin_prefetch(arow + kb + 32, 0, 3);

        // ---- NT/16 WMMAs reusing the A fragment; B frags pipelined from LDS ----
        const _Float16* brow = ldsW + (size_t)kb * NT + mloc;    // lane = N column
        v16h bcur;                                               // elements = K (table order)
#pragma unroll
        for (int s = 0; s < 16; ++s)
            bcur[s] = brow[(size_t)(((s < 8) ? s : (s + 8)) + 8 * hiK) * NT];
#pragma unroll
        for (int nt = 0; nt < NT / 16; ++nt) {
            v16h bnext = bcur;
            if (nt + 1 < NT / 16) {
#pragma unroll
                for (int s = 0; s < 16; ++s)
                    bnext[s] = brow[(size_t)(((s < 8) ? s : (s + 8)) + 8 * hiK) * NT + (nt + 1) * 16];    // fetch ahead
            }
            acc[nt] = __builtin_amdgcn_wmma_f32_16x16x32_f16(
                false, afrag, false, bcur, (short)0, acc[nt], false, false);
            bcur = bnext;
        }
    }

    // ---- epilogue: VGPR r -> M = r + 8*hiK, N = lane&15 ----
#pragma unroll
    for (int nt = 0; nt < NT / 16; ++nt) {
#pragma unroll
        for (int r = 0; r < 8; ++r) {
            int m = mb + r + 8 * hiK;
            int n = nt * 16 + mloc;
            float val = acc[nt][r] + bias[n];
            if (relu) val = fmaxf(val, 0.0f);
            size_t off = (size_t)(m / cRowsPerBatch) * (size_t)cBatchStride
                       + (size_t)(m % cRowsPerBatch) * (size_t)ldc + (size_t)n;
            C[off] = val;
        }
    }
}

// ---------------- (B,C,HW) -> (B*HW, C) transpose for the input projection ----
__global__ void k_chw_to_hwc(const float* __restrict__ f, float* __restrict__ x,
                             int C, int HW, int total)
{
    int i = blockIdx.x * blockDim.x + threadIdx.x;
    if (i >= total) return;
    int t = i % HW;
    int c = (i / HW) % C;
    int b = i / (HW * C);
    x[((size_t)b * HW + t) * C + c] = f[i];
}

// ---------------- GroupNorm over (C/32 channels, all tokens of one level) ----
__global__ __launch_bounds__(256)
void k_groupnorm(float* __restrict__ src, const float* __restrict__ gam,
                 const float* __restrict__ bet, int HW, int tokOff)
{
    __shared__ float ssum[256], ssq[256];
    int b  = blockIdx.x >> 5;   // 32 groups
    int g  = blockIdx.x & 31;
    int c0 = g * 6;
    int n  = 6 * HW;
    float s = 0.f, q = 0.f;
    for (int i = threadIdx.x; i < n; i += 256) {
        int c = c0 + i / HW;
        int t = i % HW;
        float v = src[((size_t)(b * LTOK + tokOff + t)) * D_ + c];
        s += v; q += v * v;
    }
    ssum[threadIdx.x] = s; ssq[threadIdx.x] = q;
    __syncthreads();
    for (int st = 128; st > 0; st >>= 1) {
        if (threadIdx.x < st) {
            ssum[threadIdx.x] += ssum[threadIdx.x + st];
            ssq[threadIdx.x]  += ssq[threadIdx.x + st];
        }
        __syncthreads();
    }
    float mean = ssum[0] / (float)n;
    float var  = ssq[0] / (float)n - mean * mean;
    float rstd = rsqrtf(var + 1e-5f);
    for (int i = threadIdx.x; i < n; i += 256) {
        int c = c0 + i / HW;
        int t = i % HW;
        size_t idx = ((size_t)(b * LTOK + tokOff + t)) * D_ + c;
        src[idx] = (src[idx] - mean) * rstd * gam[c] + bet[c];
    }
}

// ---------------- sine/cosine positional embedding + level embed ----------------
__global__ void k_pos(float* __restrict__ pos, const float* __restrict__ lev)
{
    int i = blockIdx.x * blockDim.x + threadIdx.x;
    if (i >= LTOK * D_) return;
    int d = i % D_;
    int t = i / D_;
    int level, H, W, tl;
    if (t < HW0_) { level = 0; H = H0_; W = W0_; tl = t; }
    else          { level = 1; H = H1_; W = W1_; tl = t - HW0_; }
    int yi = tl / W, xj = tl % W;
    float v; int dd;
    const float scale = 6.283185307179586f;
    if (d < 96) { v = ((float)yi + 0.5f) / ((float)H + 1e-6f) * scale; dd = d; }
    else        { v = ((float)xj + 0.5f) / ((float)W + 1e-6f) * scale; dd = d - 96; }
    int k = dd >> 1;
    float T = powf(10000.0f, (2.0f * (float)k) / 96.0f);
    float p = v / T;
    float val = (dd & 1) ? cosf(p) : sinf(p);
    pos[i] = val + lev[level * D_ + d];
}

// ---------------- q = out + pos (pos broadcast over batch) ----------------
__global__ void k_addpos(const float* __restrict__ src, const float* __restrict__ pos,
                         float* __restrict__ q, int total)
{
    int i = blockIdx.x * blockDim.x + threadIdx.x;
    if (i >= total) return;
    q[i] = src[i] + pos[i % (LTOK * D_)];
}

// ---------------- softmax over (NL*NP) + bilinear deformable sampling ----------------
__global__ __launch_bounds__(256)
void k_msattn(const float* __restrict__ v, const float* __restrict__ off,
              const float* __restrict__ aw, float* __restrict__ attn)
{
    int i = blockIdx.x * 256 + threadIdx.x;
    if (i >= BB * LTOK * NH_) return;
    int h = i % NH_;
    int t = (i / NH_) % LTOK;
    int b = i / (NH_ * LTOK);

    int Hq, Wq, tl;
    if (t < HW0_) { Hq = H0_; Wq = W0_; tl = t; }
    else          { Hq = H1_; Wq = W1_; tl = t - HW0_; }
    float refx = ((float)(tl % Wq) + 0.5f) / (float)Wq;
    float refy = ((float)(tl / Wq) + 0.5f) / (float)Hq;

    size_t row = (size_t)b * LTOK + t;
    const float* offp = off + row * 128 + h * 16;   // (NL,NP,2)
    const float* awp  = aw  + row * 64  + h * 8;    // (NL*NP)

    float lg[8], mx = -1e30f;
#pragma unroll
    for (int j = 0; j < 8; ++j) { lg[j] = awp[j]; mx = fmaxf(mx, lg[j]); }
    float se = 0.f;
#pragma unroll
    for (int j = 0; j < 8; ++j) { lg[j] = expf(lg[j] - mx); se += lg[j]; }
    float inv = 1.0f / se;

    float acc[HD_];
#pragma unroll
    for (int c = 0; c < HD_; ++c) acc[c] = 0.f;

    for (int l = 0; l < 2; ++l) {
        int Hl = l ? H1_ : H0_;
        int Wl = l ? W1_ : W0_;
        int to = l ? HW0_ : 0;
        for (int p = 0; p < 4; ++p) {
            float wA = lg[l * 4 + p] * inv;
            float lx = refx + offp[l * 8 + p * 2 + 0] / (float)Wl;
            float ly = refy + offp[l * 8 + p * 2 + 1] / (float)Hl;
            float x = lx * (float)Wl - 0.5f;
            float y = ly * (float)Hl - 0.5f;
            float x0f = floorf(x), y0f = floorf(y);
            float wx = x - x0f, wy = y - y0f;
            int x0 = (int)x0f, y0 = (int)y0f;
            int   cx[4] = { x0, x0 + 1, x0,     x0 + 1 };
            int   cy[4] = { y0, y0,     y0 + 1, y0 + 1 };
            float cw[4] = { (1 - wx) * (1 - wy), wx * (1 - wy),
                            (1 - wx) * wy,       wx * wy };
            for (int q = 0; q < 4; ++q) {
                int xc = cx[q], yc = cy[q];
                if (xc < 0 || xc > Wl - 1 || yc < 0 || yc > Hl - 1) continue;
                const float* vp = v + ((size_t)b * LTOK + to + yc * Wl + xc) * D_ + h * HD_;
                float w = wA * cw[q];
#pragma unroll
                for (int c = 0; c < HD_; ++c) acc[c] += w * vp[c];
            }
        }
    }
    float* op = attn + row * D_ + h * HD_;
#pragma unroll
    for (int c = 0; c < HD_; ++c) op[c] = acc[c];
}

// ---------------- LayerNorm(out + delta) in place (one wave per row) ----------------
__global__ __launch_bounds__(256)
void k_ln(float* __restrict__ out, const float* __restrict__ delta,
          const float* __restrict__ gam, const float* __restrict__ bet, int rows)
{
    int w    = (blockIdx.x * 256 + threadIdx.x) >> 5;
    int lane = threadIdx.x & 31;
    if (w >= rows) return;
    size_t base = (size_t)w * D_;
    float x[6];
    float s = 0.f;
#pragma unroll
    for (int i = 0; i < 6; ++i) {
        x[i] = out[base + lane + 32 * i] + delta[base + lane + 32 * i];
        s += x[i];
    }
#pragma unroll
    for (int o = 16; o > 0; o >>= 1) s += __shfl_xor(s, o, 32);
    float mean = s / (float)D_;
    float q = 0.f;
#pragma unroll
    for (int i = 0; i < 6; ++i) { float dd = x[i] - mean; q += dd * dd; }
#pragma unroll
    for (int o = 16; o > 0; o >>= 1) q += __shfl_xor(q, o, 32);
    float rstd = rsqrtf(q / (float)D_ + 1e-5f);
#pragma unroll
    for (int i = 0; i < 6; ++i) {
        int c = lane + 32 * i;
        out[base + c] = (x[i] - mean) * rstd * gam[c] + bet[c];
    }
}

// ---------------- token-major -> (B,C,H,W) for both outputs ----------------
__global__ void k_write(const float* __restrict__ src, float* __restrict__ out, int total)
{
    int i = blockIdx.x * blockDim.x + threadIdx.x;
    if (i >= total) return;
    const int n0 = BB * D_ * HW0_;
    float val;
    if (i < n0) {
        int b = i / (D_ * HW0_);
        int r = i % (D_ * HW0_);
        int d = r / HW0_;
        int t = r % HW0_;
        val = src[((size_t)b * LTOK + t) * D_ + d];
    } else {
        int j = i - n0;
        int b = j / (D_ * HW1_);
        int r = j % (D_ * HW1_);
        int d = r / HW1_;
        int t = r % HW1_;
        val = src[((size_t)b * LTOK + HW0_ + t) * D_ + d];
    }
    out[i] = val;
}

// ---------------------------------------------------------------------------
extern "C" void kernel_launch(void* const* d_in, const int* in_sizes, int n_in,
                              void* d_out, int out_size, void* d_ws, size_t ws_size,
                              hipStream_t stream)
{
    (void)in_sizes; (void)n_in; (void)ws_size;
    const float* feat0      = (const float*)d_in[0];
    const float* feat1      = (const float*)d_in[1];
    // d_in[2] (imgs) unused by the reference
    const float* proj_w0    = (const float*)d_in[3];
    const float* proj_b0    = (const float*)d_in[4];
    const float* gn_g0      = (const float*)d_in[5];
    const float* gn_b0      = (const float*)d_in[6];
    const float* proj_w1    = (const float*)d_in[7];
    const float* proj_b1    = (const float*)d_in[8];
    const float* gn_g1      = (const float*)d_in[9];
    const float* gn_b1      = (const float*)d_in[10];
    const float* level_emb  = (const float*)d_in[11];
    const float* so_W       = (const float*)d_in[12];
    const float* so_b       = (const float*)d_in[13];
    const float* aw_W       = (const float*)d_in[14];
    const float* aw_b       = (const float*)d_in[15];
    const float* vp_W       = (const float*)d_in[16];
    const float* vp_b       = (const float*)d_in[17];
    const float* op_W       = (const float*)d_in[18];
    const float* op_b       = (const float*)d_in[19];
    const float* ln1_g      = (const float*)d_in[20];
    const float* ln1_b      = (const float*)d_in[21];
    const float* l1_W       = (const float*)d_in[22];
    const float* l1_b       = (const float*)d_in[23];
    const float* l2_W       = (const float*)d_in[24];
    const float* l2_b       = (const float*)d_in[25];
    const float* ln2_g      = (const float*)d_in[26];
    const float* ln2_b      = (const float*)d_in[27];

    // ------- workspace layout (floats) -------
    float* ws   = (float*)d_ws;
    const size_t BLD = (size_t)BB * LTOK * D_;          // 4,423,680
    float* srcb  = ws;                                  // running "out"
    float* posb  = srcb  + BLD;                         // (L, D)
    float* vbuf  = posb  + (size_t)LTOK * D_;           // value projection
    float* attnb = vbuf  + BLD;                         // sampled attention
    float* tmpb  = attnb + BLD;                         // op / ffn2 output
    float* offb  = tmpb  + BLD;                         // (B*L, 128)
    float* awb   = offb  + (size_t)BB * LTOK * 128;     // (B*L, 64)
    float* bigb  = awb   + (size_t)BB * LTOK * 64;      // (B*L, 256): q / h1 / transpose scratch

    const int M = BB * LTOK;    // 23040 rows for encoder GEMMs (M/128 = 180)

    // ------- input projection + group norm, level 0 -------
    {
        int total = BB * D_ * HW0_;
        k_chw_to_hwc<<<(total + 255) / 256, 256, 0, stream>>>(feat0, bigb, D_, HW0_, total);
        k_gemm<192><<<dim3(BB * HW0_ / 128), 256, D_ * 192 * 2, stream>>>(
            bigb, D_, proj_w0, D_, /*transB=*/1, proj_b0,
            srcb, D_, HW0_, (long long)LTOK * D_, BB * HW0_, D_, 0);
        k_groupnorm<<<BB * 32, 256, 0, stream>>>(srcb, gn_g0, gn_b0, HW0_, 0);
    }
    // ------- input projection + group norm, level 1 -------
    {
        int total = BB * D_ * HW1_;
        k_chw_to_hwc<<<(total + 255) / 256, 256, 0, stream>>>(feat1, bigb, D_, HW1_, total);
        k_gemm<192><<<dim3(BB * HW1_ / 128), 256, D_ * 192 * 2, stream>>>(
            bigb, D_, proj_w1, D_, /*transB=*/1, proj_b1,
            srcb + (size_t)HW0_ * D_, D_, HW1_, (long long)LTOK * D_, BB * HW1_, D_, 0);
        k_groupnorm<<<BB * 32, 256, 0, stream>>>(srcb, gn_g1, gn_b1, HW1_, HW0_);
    }
    // ------- positional embedding -------
    k_pos<<<(LTOK * D_ + 255) / 256, 256, 0, stream>>>(posb, level_emb);

    // ------- encoder layers -------
    for (int i = 0; i < NLAYERS; ++i) {
        const float* soW  = so_W + (size_t)i * D_ * 128;
        const float* sob  = so_b + (size_t)i * 128;
        const float* awW  = aw_W + (size_t)i * D_ * 64;
        const float* awbb = aw_b + (size_t)i * 64;
        const float* vpW  = vp_W + (size_t)i * D_ * D_;
        const float* vpb  = vp_b + (size_t)i * D_;
        const float* opW  = op_W + (size_t)i * D_ * D_;
        const float* opb  = op_b + (size_t)i * D_;
        const float* l1W  = l1_W + (size_t)i * D_ * 256;
        const float* l1b  = l1_b + (size_t)i * 256;
        const float* l2W  = l2_W + (size_t)i * 256 * D_;
        const float* l2b  = l2_b + (size_t)i * D_;
        const float* g1   = ln1_g + (size_t)i * D_;
        const float* b1   = ln1_b + (size_t)i * D_;
        const float* g2   = ln2_g + (size_t)i * D_;
        const float* b2   = ln2_b + (size_t)i * D_;

        // q = out + pos
        k_addpos<<<(BB * LTOK * D_ + 255) / 256, 256, 0, stream>>>(srcb, posb, bigb, BB * LTOK * D_);
        // sampling offsets: (B*L,192) x (192,128)
        k_gemm<128><<<dim3(M / 128), 256, D_ * 128 * 2, stream>>>(
            bigb, D_, soW, 128, 0, sob, offb, 128, M, 0, M, D_, 0);
        // attention-weight logits: (B*L,192) x (192,64)
        k_gemm<64><<<dim3(M / 128), 256, D_ * 64 * 2, stream>>>(
            bigb, D_, awW, 64, 0, awbb, awb, 64, M, 0, M, D_, 0);
        // value projection: (B*L,192) x (192,192)
        k_gemm<192><<<dim3(M / 128), 256, D_ * 192 * 2, stream>>>(
            srcb, D_, vpW, D_, 0, vpb, vbuf, D_, M, 0, M, D_, 0);
        // softmax + deformable bilinear sampling
        k_msattn<<<(BB * LTOK * NH_ + 255) / 256, 256, 0, stream>>>(vbuf, offb, awb, attnb);
        // output projection
        k_gemm<192><<<dim3(M / 128), 256, D_ * 192 * 2, stream>>>(
            attnb, D_, opW, D_, 0, opb, tmpb, D_, M, 0, M, D_, 0);
        // out = LN(out + attn)
        k_ln<<<(M * 32 + 255) / 256, 256, 0, stream>>>(srcb, tmpb, g1, b1, M);
        // FFN: h1 = relu(out @ l1 + b)
        k_gemm<256><<<dim3(M / 128), 256, D_ * 256 * 2, stream>>>(
            srcb, D_, l1W, 256, 0, l1b, bigb, 256, M, 0, M, D_, 1);
        // ff = h1 @ l2 + b
        k_gemm<192><<<dim3(M / 128), 256, 256 * 192 * 2, stream>>>(
            bigb, 256, l2W, D_, 0, l2b, tmpb, D_, M, 0, M, 256, 0);
        // out = LN(out + ff)
        k_ln<<<(M * 32 + 255) / 256, 256, 0, stream>>>(srcb, tmpb, g2, b2, M);
    }

    // ------- write outputs (m0 then m1, channel-major) -------
    k_write<<<(out_size + 255) / 256, 256, 0, stream>>>(srcb, (float*)d_out, out_size);
}
